// CausalSelfAttention_68169720922226
// MI455X (gfx1250) — compile-verified, matrix operand order checked by reading, not run on hardware
//
#include <hip/hip_runtime.h>

// ---------------------------------------------------------------------------
// CDNA5 (gfx1250) causal self-attention, bf16 WMMA pipeline.
//   B=4, T=2048, C=768, H=12, D=64
// ---------------------------------------------------------------------------

#define B_  4
#define T_  2048
#define C_  768
#define H_  12
#define D_  64
#define TC_ (3 * C_)   // 2304

typedef __attribute__((ext_vector_type(16))) __bf16 v16bf;
typedef __attribute__((ext_vector_type(8)))  float  v8f;
typedef int v4i_ __attribute__((ext_vector_type(4)));

// ---- gfx1250 async global->LDS path (guarded; falls back to sync copy) ----
#if defined(__AMDGCN__) && __has_builtin(__builtin_amdgcn_global_load_async_to_lds_b128)
#define HAVE_ASYNC_LDS 1
typedef __attribute__((address_space(1))) v4i_ as1_v4i;
typedef __attribute__((address_space(3))) v4i_ as3_v4i;
#else
#define HAVE_ASYNC_LDS 0
#endif

__device__ __forceinline__ void async_copy16(const void* g, void* l) {
#if HAVE_ASYNC_LDS
    __builtin_amdgcn_global_load_async_to_lds_b128(
        (as1_v4i*)g, (as3_v4i*)l, 0, 0);
#else
    *(uint4*)l = *(const uint4*)g;
#endif
}

__device__ __forceinline__ void async_wait0() {
#if HAVE_ASYNC_LDS
#if __has_builtin(__builtin_amdgcn_s_wait_asynccnt)
    __builtin_amdgcn_s_wait_asynccnt(0);
#else
    asm volatile("s_wait_asynccnt 0" ::: "memory");
#endif
#endif
}

__device__ __forceinline__ v16bf make_frag(uint4 lo, uint4 hi) {
    union { uint4 q[2]; v16bf v; } u;
    u.q[0] = lo; u.q[1] = hi;
    return u.v;
}

// Load one 16x32 bf16 fragment in the A-operand layout from a row-major
// [rows][stride] bf16 tile. Lane L: m = L%16, K-half = (L/16)*8.
//   elements 0-7 : K in [kh, kh+8)        -> one b128
//   elements 8-15: K in [16+kh, 16+kh+8)  -> one b128
// stride*2 must be a multiple of 16 bytes (40 and 72 both work).
__device__ __forceinline__ v16bf load_frag(const __bf16* p0, int stride, int lane) {
    const int m  = lane & 15;
    const int kh = (lane >> 4) << 3;
    const __bf16* p = p0 + m * stride + kh;
    return make_frag(*(const uint4*)p, *(const uint4*)(p + 16));
}

__device__ __forceinline__ v8f wmma_bf16(v16bf a, v16bf b, v8f c) {
    return __builtin_amdgcn_wmma_f32_16x16x32_bf16(
        false, a, false, b, (short)0, c, false, false);
}

// ---------------------------------------------------------------------------
// Kernel 1: f32 -> bf16 elementwise
// ---------------------------------------------------------------------------
__global__ void cvt_f32_to_bf16(const float* __restrict__ in,
                                __bf16* __restrict__ out, int n) {
    for (int i = blockIdx.x * blockDim.x + threadIdx.x; i < n;
         i += gridDim.x * blockDim.x)
        out[i] = (__bf16)in[i];
}

// ---------------------------------------------------------------------------
// Kernel 2: f32 [rows][cols] -> bf16 transposed [cols][rows]
// ---------------------------------------------------------------------------
__global__ void transpose_cvt(const float* __restrict__ in,
                              __bf16* __restrict__ out, int rows, int cols) {
    int total = rows * cols;
    for (int i = blockIdx.x * blockDim.x + threadIdx.x; i < total;
         i += gridDim.x * blockDim.x) {
        int k = i % rows;   // out is [cols][rows]
        int n = i / rows;
        out[i] = (__bf16)in[(size_t)k * cols + n];
    }
}

// ---------------------------------------------------------------------------
// Kernel 3: bf16 WMMA GEMM   out[M][N] = A[M][K] * Bt[N][K]^T + bias[N]
// Block tile 128x128, 8 waves, wave tile 32x64 (2x4 accum frags).
// Double-buffered LDS (TK=32) fed by async global->LDS DMA; stage k+1 is in
// flight while stage k runs on the matrix cores.
// M,N multiples of 128, K multiple of 32 (true for all uses here).
// ---------------------------------------------------------------------------
template <bool OUT_BF16>
__global__ __launch_bounds__(256) void gemm_bf16(
    const __bf16* __restrict__ A, const __bf16* __restrict__ Bt,
    const float* __restrict__ bias, void* __restrict__ out,
    int M, int N, int K) {
    constexpr int TM = 128, TN = 128, TK = 32, LDT = 40;
    __shared__ alignas(16) __bf16 As[2][TM][LDT];   // 2 x 10 KB
    __shared__ alignas(16) __bf16 Bs[2][TN][LDT];   // 2 x 10 KB

    const int tid  = threadIdx.x;
    const int lane = tid & 31;
    const int w    = tid >> 5;
    const int wm   = w & 3;   // 4 M-strips of 32
    const int wn   = w >> 2;  // 2 N-strips of 64

    const int ntn = N / TN;
    const int mb  = blockIdx.x / ntn;
    const int nb  = blockIdx.x % ntn;

    const __bf16* Ab = A + (size_t)mb * TM * K;
    const __bf16* Bb = Bt + (size_t)nb * TN * K;

    // 128x32 tile = 512 16-byte chunks -> 2 chunks per thread per matrix.
    auto issue = [&](int k0, int buf) {
#pragma unroll
        for (int i = 0; i < 2; ++i) {
            int u   = tid + 256 * i;      // 0..511
            int row = u >> 2;
            int cg  = (u & 3) * 8;
            async_copy16(&Ab[(size_t)row * K + k0 + cg], &As[buf][row][cg]);
            async_copy16(&Bb[(size_t)row * K + k0 + cg], &Bs[buf][row][cg]);
        }
    };

    v8f acc[2][4] = {};
    const int nk = K / TK;

    issue(0, 0);
    for (int it = 0; it < nk; ++it) {
        async_wait0();        // stage `it` landed in LDS
        __syncthreads();      // visible to all waves; prior reads of buf^1 done
        if (it + 1 < nk) issue((it + 1) * TK, (it + 1) & 1);

        const int buf = it & 1;
        v16bf a[2], b[4];
#pragma unroll
        for (int i = 0; i < 2; ++i)
            a[i] = load_frag(&As[buf][wm * 32 + i * 16][0], LDT, lane);
#pragma unroll
        for (int j = 0; j < 4; ++j)
            b[j] = load_frag(&Bs[buf][wn * 64 + j * 16][0], LDT, lane);
#pragma unroll
        for (int i = 0; i < 2; ++i)
#pragma unroll
            for (int j = 0; j < 4; ++j)
                acc[i][j] = wmma_bf16(a[i], b[j], acc[i][j]);
    }

    // Epilogue. C-layout: element v -> row = v + 8*(lane/16), col = lane%16.
    const int nlane = lane & 15;
    const int half  = lane >> 4;
    const int mBase = mb * TM + wm * 32;
    const int nBase = nb * TN + wn * 64;
#pragma unroll
    for (int i = 0; i < 2; ++i)
#pragma unroll
        for (int j = 0; j < 4; ++j) {
            int ncol = nBase + j * 16 + nlane;
            float bv = bias[ncol];
#pragma unroll
            for (int v = 0; v < 8; ++v) {
                int mrow = mBase + i * 16 + v + 8 * half;
                float val = acc[i][j][v] + bv;
                if (OUT_BF16)
                    ((__bf16*)out)[(size_t)mrow * N + ncol] = (__bf16)val;
                else
                    ((float*)out)[(size_t)mrow * N + ncol] = val;
            }
        }
}

// ---------------------------------------------------------------------------
// Kernel 4: flash attention (causal, online softmax), bf16 WMMA.
// Grid: (T/128) * B * H blocks of 256 threads (8 waves).
// Each wave owns 16 query rows; K/V tiles of 64 keys staged in LDS.
// K tile arrives via async global->LDS DMA; V is transposed through VGPRs.
// qkv layout: [B][T][3C], q at h*64, k at C+h*64, v at 2C+h*64.
// Output y: bf16 [B*T][C].
// ---------------------------------------------------------------------------
__global__ __launch_bounds__(256) void flash_attn(
    const __bf16* __restrict__ qkv, __bf16* __restrict__ y) {
    constexpr int QB = 128, KB = 64, LDT = 72;
    __shared__ alignas(16) __bf16 Ks[KB][LDT];       // [key][d]
    __shared__ alignas(16) __bf16 Vt[D_][LDT];       // [d][key]  (transposed)
    __shared__ alignas(16) __bf16 Ps[8][16][LDT];    // per-wave P buffer

    const int tid  = threadIdx.x;
    const int lane = tid & 31;
    const int w    = tid >> 5;

    const int nQB = T_ / QB;                 // 16
    const int bh  = blockIdx.x / nQB;
    const int qb  = blockIdx.x % nQB;
    const int b   = bh / H_;
    const int h   = bh % H_;

    const size_t rs    = TC_;                // qkv row stride (elements)
    const __bf16* base = qkv + (size_t)b * T_ * rs;
    const int qRow0    = qb * QB + w * 16;   // wave's first query row

    // Q strip as two A-fragments (D = 64 -> 2 K-steps of 32), loaded once.
    v16bf qf[2];
    {
        const int m  = lane & 15;
        const int kh = (lane >> 4) << 3;
        const __bf16* qp = base + (size_t)(qRow0 + m) * rs + h * D_;
#pragma unroll
        for (int f = 0; f < 2; ++f)
            qf[f] = make_frag(*(const uint4*)(qp + f * 32 + kh),
                              *(const uint4*)(qp + f * 32 + kh + 16));
    }

    v8f   o[4] = {};
    float mrow[8], lrow[8];
#pragma unroll
    for (int v = 0; v < 8; ++v) { mrow[v] = -1e30f; lrow[v] = 0.0f; }

    const int   nlane = lane & 15;
    const int   half  = lane >> 4;
    const float scale = 0.125f;              // 1/sqrt(64)
    const int   kEnd  = qb * QB + QB;        // causal: keys < block end

    for (int kb = 0; kb < kEnd; kb += KB) {
        __syncthreads();   // previous tile fully consumed
        // K tile (natural [key][d]) via async DMA; 512 chunks -> 2/thread.
#pragma unroll
        for (int i = 0; i < 2; ++i) {
            int u = tid + 256 * i;           // 0..511
            int row = u >> 3, cg = (u & 7) * 8;
            async_copy16(&base[(size_t)(kb + row) * rs + C_ + h * D_ + cg],
                         &Ks[row][cg]);
        }
        // V tile, transposed into [d][key] through VGPRs.
#pragma unroll
        for (int i = 0; i < 2; ++i) {
            int u = tid + 256 * i;
            int row = u >> 3, cg = (u & 7) * 8;
            uint4 vv =
                *(const uint4*)&base[(size_t)(kb + row) * rs + 2 * C_ + h * D_ + cg];
            const __bf16* e = (const __bf16*)&vv;
#pragma unroll
            for (int j = 0; j < 8; ++j) Vt[cg + j][row] = e[j];
        }
        async_wait0();
        __syncthreads();

        // S = (Q * K^T): 4 n-frags (16 keys each) x 2 k-steps over D.
        v8f s[4];
#pragma unroll
        for (int f = 0; f < 4; ++f) {
            v8f c = {};
#pragma unroll
            for (int kk = 0; kk < 2; ++kk)
                c = wmma_bf16(qf[kk], load_frag(&Ks[f * 16][kk * 32], LDT, lane), c);
            s[f] = c;
        }

        // Scale + causal mask + online softmax.
        float mNew[8], corr[8];
#pragma unroll
        for (int v = 0; v < 8; ++v) {
            const int q = qRow0 + v + 8 * half;
            float mx = -1e30f;
#pragma unroll
            for (int f = 0; f < 4; ++f) {
                int key = kb + f * 16 + nlane;
                float sv = (key <= q) ? s[f][v] * scale : -1e30f;
                s[f][v] = sv;
                mx = fmaxf(mx, sv);
            }
#pragma unroll
            for (int off = 1; off < 16; off <<= 1)
                mx = fmaxf(mx, __shfl_xor(mx, off, 32));
            mNew[v] = fmaxf(mrow[v], mx);
            corr[v] = __expf(mrow[v] - mNew[v]);
            mrow[v] = mNew[v];
        }
#pragma unroll
        for (int v = 0; v < 8; ++v) {
            float sum = 0.0f;
#pragma unroll
            for (int f = 0; f < 4; ++f) {
                float p = __expf(s[f][v] - mNew[v]);
                s[f][v] = p;
                sum += p;
            }
#pragma unroll
            for (int off = 1; off < 16; off <<= 1)
                sum += __shfl_xor(sum, off, 32);
            lrow[v] = lrow[v] * corr[v] + sum;
        }
#pragma unroll
        for (int f = 0; f < 4; ++f)
#pragma unroll
            for (int v = 0; v < 8; ++v) o[f][v] *= corr[v];

        // P (C-layout) -> per-wave LDS (row-major) -> A-fragments.
        // LDS ops are in-order within a wave, so no block barrier needed.
#pragma unroll
        for (int f = 0; f < 4; ++f)
#pragma unroll
            for (int v = 0; v < 8; ++v)
                Ps[w][v + 8 * half][f * 16 + nlane] = (__bf16)s[f][v];

        // O += P * V  (V transposed tile gives B operand via A-style load)
#pragma unroll
        for (int kk = 0; kk < 2; ++kk) {
            v16bf pa = load_frag(&Ps[w][0][kk * 32], LDT, lane);
#pragma unroll
            for (int f = 0; f < 4; ++f)
                o[f] = wmma_bf16(pa, load_frag(&Vt[f * 16][kk * 32], LDT, lane), o[f]);
        }
    }

    // y[b*T + q][h*64 + d] = O / l
#pragma unroll
    for (int f = 0; f < 4; ++f)
#pragma unroll
        for (int v = 0; v < 8; ++v) {
            int qrow = qRow0 + v + 8 * half;
            float val = o[f][v] / lrow[v];
            y[(size_t)(b * T_ + qrow) * C_ + h * D_ + f * 16 + nlane] = (__bf16)val;
        }
}

// ---------------------------------------------------------------------------
// Host-side launcher
// ---------------------------------------------------------------------------
extern "C" void kernel_launch(void* const* d_in, const int* in_sizes, int n_in,
                              void* d_out, int out_size, void* d_ws, size_t ws_size,
                              hipStream_t stream) {
    const float* x      = (const float*)d_in[0];   // [B,T,C]
    const float* w_attn = (const float*)d_in[1];   // [C,3C]
    const float* b_attn = (const float*)d_in[2];   // [3C]
    const float* w_proj = (const float*)d_in[3];   // [C,C]
    const float* b_proj = (const float*)d_in[4];   // [C]
    float* out = (float*)d_out;

    const int M = B_ * T_;                         // 8192

    // Workspace layout (all sizes already 256B-aligned)
    char* p = (char*)d_ws;
    __bf16* x_bf   = (__bf16*)p; p += (size_t)M * C_ * 2;        // 12.6 MB
    __bf16* wt_att = (__bf16*)p; p += (size_t)TC_ * C_ * 2;      //  3.5 MB  [3C][C]
    __bf16* wt_prj = (__bf16*)p; p += (size_t)C_ * C_ * 2;       //  1.2 MB  [C][C]
    __bf16* qkv_bf = (__bf16*)p; p += (size_t)M * TC_ * 2;       // 37.7 MB
    __bf16* y_bf   = (__bf16*)p;                                 // 12.6 MB

    // 1) convert x to bf16
    cvt_f32_to_bf16<<<4096, 256, 0, stream>>>(x, x_bf, M * C_);
    // 2) transpose+convert weights to [N][K] bf16
    transpose_cvt<<<2048, 256, 0, stream>>>(w_attn, wt_att, C_, TC_);
    transpose_cvt<<<1024, 256, 0, stream>>>(w_proj, wt_prj, C_, C_);

    // 3) qkv = x @ w_attn + b_attn  (bf16 out)
    gemm_bf16<true><<<(M / 128) * (TC_ / 128), 256, 0, stream>>>(
        x_bf, wt_att, b_attn, qkv_bf, M, TC_, C_);

    // 4) flash attention -> y (bf16)
    flash_attn<<<(T_ / 128) * B_ * H_, 256, 0, stream>>>(qkv_bf, y_bf);

    // 5) out = y @ w_proj + b_proj  (f32 out)
    gemm_bf16<false><<<(M / 128) * (C_ / 128), 256, 0, stream>>>(
        y_bf, wt_prj, b_proj, out, M, C_, C_);
}
